// AttentionBlock_73693048865276
// MI455X (gfx1250) — compile-verified, matrix operand order checked by reading, not run on hardware
//
#include <hip/hip_runtime.h>
#include <hip/hip_bf16.h>

// ---------------------------------------------------------------------------
// AttentionBlock on gfx1250 (MI455X): GroupNorm -> QKV GEMM -> fused attention
// (LDS-resident 64x1024 score tile; 320KB WGP LDS) -> proj GEMM + residual.
// All matmuls: v_wmma_f32_16x16x32_bf16. All global->LDS staging uses the
// CDNA5 async-to-LDS engine (ASYNCcnt) with double buffering: stage k+1 is
// issued, then s_wait_asynccnt <= (ops just issued) -- async ops retire in
// order per wave, so tile k is guaranteed resident while k+1 is in flight.
// Layouts picked so every staging copy is contiguous on BOTH sides:
//   XN, AO : [b][L][C]   (K-major for GEMM B tiles)
//   Q, K   : [b][h][L][D]
//   V      : [b][h][D][L]
// ---------------------------------------------------------------------------

typedef __attribute__((ext_vector_type(16))) __bf16 v16bf;
typedef __attribute__((ext_vector_type(8)))  float  v8f;

#define B_  16
#define C_  512
#define L_  1024
#define NH_ 4
#define HD_ 128
#define NG_ 8
#define CPG_ 64

// --- CDNA5 async global->LDS copy (ASYNCcnt) -------------------------------
#define USE_ASYNC_COPY 1

__device__ __forceinline__ void cp_b128(void* lds, const void* gsrc) {
#if USE_ASYNC_COPY
    asm volatile("global_load_async_to_lds_b128 %0, %1, off"
                 :: "v"((unsigned)(unsigned long long)lds), "v"(gsrc)
                 : "memory");
#else
    *(uint4*)lds = *(const uint4*)gsrc;
#endif
}
#if USE_ASYNC_COPY
#define CP_WAIT(n) asm volatile("s_wait_asynccnt " #n ::: "memory")
#else
#define CP_WAIT(n) do {} while (0)
#endif

// --- 16-bit WMMA fragment K mapping (ISA 7.12.2, 16-bit A 16x32) -----------
__device__ __forceinline__ int frag_kbase(int lane, int j) {
    int kb = (j < 4) ? (2 * j) : (16 + 2 * (j - 4));
    if (lane & 16) kb += 8;
    return kb;
}

// Fragment load: LDS tile laid out [row][k] pairs-contiguous, even stride,
// even col0 -> dword-aligned ds loads (compiler fuses into ds_load_b128 x2).
__device__ __forceinline__ v16bf ld_frag(const __bf16* tile, int row, int stride,
                                         int col0, int lane) {
    union { v16bf v; unsigned int u[8]; } f;
    const unsigned int* p = (const unsigned int*)(tile + row * stride + col0);
#pragma unroll
    for (int j = 0; j < 8; ++j) f.u[j] = p[frag_kbase(lane, j) >> 1];
    return f.v;
}

__device__ __forceinline__ v8f wmma_bf16(v16bf a, v16bf b, v8f c) {
    return __builtin_amdgcn_wmma_f32_16x16x32_bf16(false, a, false, b,
                                                   (short)0, c, false, false);
}

union Pack8 { uint4 u; __bf16 h[8]; };

// ---------------------------------------------------------------------------
// Kernel 1: fp32 -> bf16 elementwise (weights)
// ---------------------------------------------------------------------------
__global__ void f32_to_bf16_kernel(const float* __restrict__ in,
                                   __bf16* __restrict__ out, int n) {
    int i = blockIdx.x * blockDim.x + threadIdx.x;
    if (i < n) out[i] = (__bf16)in[i];
}

// ---------------------------------------------------------------------------
// Kernel 2: GroupNorm per (b,g); emit bf16 XN in K-major [b][L][C] layout
// with 16B-vectorized stores (8 channels per uint4).
// ---------------------------------------------------------------------------
__global__ void groupnorm_kernel(const float* __restrict__ x,
                                 const float* __restrict__ gamma,
                                 const float* __restrict__ beta,
                                 __bf16* __restrict__ xnt) {
    const int g = blockIdx.x, b = blockIdx.y, tid = threadIdx.x;
    const float* xp = x + (size_t)(b * NG_ + g) * CPG_ * L_;
    __shared__ float s1[256], s2[256];
    float sum = 0.f, sq = 0.f;
    for (int i = tid; i < CPG_ * L_; i += 256) {
        float v = xp[i]; sum += v; sq += v * v;
    }
    s1[tid] = sum; s2[tid] = sq;
    __syncthreads();
    for (int s = 128; s > 0; s >>= 1) {
        if (tid < s) { s1[tid] += s1[tid + s]; s2[tid] += s2[tid + s]; }
        __syncthreads();
    }
    const float mean = s1[0] * (1.0f / (CPG_ * L_));
    const float var  = s2[0] * (1.0f / (CPG_ * L_)) - mean * mean;
    const float inv  = rsqrtf(var + 1e-5f);

    for (int lb = 0; lb < 4; ++lb) {
        int l = lb * 256 + tid;
#pragma unroll
        for (int c8 = 0; c8 < 8; ++c8) {
            Pack8 pk;
#pragma unroll
            for (int j = 0; j < 8; ++j) {
                int cl = c8 * 8 + j;
                float v = (xp[(size_t)cl * L_ + l] - mean) * inv
                          * gamma[g * CPG_ + cl] + beta[g * CPG_ + cl];
                pk.h[j] = (__bf16)v;
            }
            *(uint4*)&xnt[((size_t)b * L_ + l) * C_ + g * CPG_ + c8 * 8] = pk.u;
        }
    }
}

// ---------------------------------------------------------------------------
// Kernel 3/5: GEMM  out[M][L] = W[M][512] @ X[L][512]^T (+bias, epilogues)
// Block tile 64x128, 8 waves, wave = 16x64 (4 accumulators, shared A frag),
// K step 32, double-buffered async LDS staging (3 async ops/wave/stage).
//   flag 0: Q/K -> [b][h][L][D] via LDS transpose + b128 stores;
//           V   -> [b][h][D][L] direct coalesced stores
//   flag 1: fp32 out [b][C][L] + bias + residual
// ---------------------------------------------------------------------------
#define AT_STRIDE 40          // 80B rows: 16B aligned, conflict-free mod 64
#define TT_STRIDE 72
#define AB_ELEMS (64 * AT_STRIDE)     // 2560
#define BB_ELEMS (128 * AT_STRIDE)    // 5120

__global__ void gemm_kernel(const __bf16* __restrict__ W,
                            const __bf16* __restrict__ Xall,   // [b][L][C]
                            const float*  __restrict__ bias,
                            __bf16* __restrict__ Qp, __bf16* __restrict__ Kp,
                            __bf16* __restrict__ Vp,
                            float* __restrict__ out,
                            const float* __restrict__ x_in,
                            int flag) {
    __shared__ __bf16 smem[2 * AB_ELEMS + 2 * BB_ELEMS];   // 15360 elems

    const int n0 = blockIdx.x * 128;
    const int m0 = blockIdx.y * 64;
    const int b  = blockIdx.z;
    const int tid = threadIdx.x, lane = tid & 31, wv = tid >> 5;
    const int mrow = (wv & 3) * 16;
    const int ncol = (wv >> 2) * 64;
    const __bf16* X = Xall + (size_t)b * L_ * C_;

    v8f acc[4];
#pragma unroll
    for (int t = 0; t < 4; ++t) acc[t] = (v8f){0.f,0.f,0.f,0.f,0.f,0.f,0.f,0.f};

    const int am = tid >> 2, ak = (tid & 3) * 8;
    const int bn1 = (tid + 256) >> 2, bk1 = ((tid + 256) & 3) * 8;

    auto stage = [&](int buf, int k0) {
        __bf16* Ab = smem + buf * AB_ELEMS;
        __bf16* Bb = smem + 2 * AB_ELEMS + buf * BB_ELEMS;
        cp_b128(&Ab[am * AT_STRIDE + ak],   &W[(size_t)(m0 + am) * C_ + k0 + ak]);
        cp_b128(&Bb[am * AT_STRIDE + ak],   &X[(size_t)(n0 + am) * C_ + k0 + ak]);
        cp_b128(&Bb[bn1 * AT_STRIDE + bk1], &X[(size_t)(n0 + bn1) * C_ + k0 + bk1]);
    };

    stage(0, 0);
    for (int it = 0; it < C_ / 32; ++it) {
        const int cur = it & 1;
        if (it + 1 < C_ / 32) { stage(1 - cur, (it + 1) * 32); CP_WAIT(0x3); }
        else                  { CP_WAIT(0x0); }
        __syncthreads();
        const __bf16* Ab = smem + cur * AB_ELEMS;
        const __bf16* Bb = smem + 2 * AB_ELEMS + cur * BB_ELEMS;
        v16bf a = ld_frag(Ab, mrow + (lane & 15), AT_STRIDE, 0, lane);
#pragma unroll
        for (int t = 0; t < 4; ++t) {
            v16bf bt = ld_frag(Bb, ncol + t * 16 + (lane & 15), AT_STRIDE, 0, lane);
            acc[t] = wmma_bf16(a, bt, acc[t]);
        }
        __syncthreads();
    }

    // ---- epilogue: C/D layout -> n = lane&15, m = r + 8*(lane>=16) --------
    if (flag == 1) {
#pragma unroll
        for (int t = 0; t < 4; ++t) {
#pragma unroll
            for (int r = 0; r < 8; ++r) {
                int m  = m0 + mrow + r + ((lane & 16) ? 8 : 0);
                int nA = n0 + ncol + t * 16 + (lane & 15);
                size_t base = (size_t)(b * C_ + m) * L_;
                out[base + nA] = acc[t][r] + bias[m] + x_in[base + nA];
            }
        }
        return;
    }

    const int part = m0 >> 9;             // 0=q 1=k 2=v (uniform per block)
    const int cb   = m0 & (C_ - 1);
    const int hh   = cb >> 7;
    const int d0   = cb & (HD_ - 1);      // 0 or 64 (uniform)

    if (part == 2) {
        // V: [b][h][d][l] -> direct, lanes contiguous along l
#pragma unroll
        for (int t = 0; t < 4; ++t) {
#pragma unroll
            for (int r = 0; r < 8; ++r) {
                int ml = mrow + r + ((lane & 16) ? 8 : 0);
                int nA = n0 + ncol + t * 16 + (lane & 15);
                size_t base = ((size_t)(b * NH_ + hh) * HD_ + d0 + ml) * L_;
                Vp[base + nA] = (__bf16)(acc[t][r] + bias[m0 + ml]);
            }
        }
    } else {
        // Q/K: [b][h][l][d] -> transpose 128x64 tile through LDS, b128 out
        __bf16* T = smem;                 // 128*72 = 9216 <= 15360 elems
#pragma unroll
        for (int t = 0; t < 4; ++t) {
#pragma unroll
            for (int r = 0; r < 8; ++r) {
                int ml = mrow + r + ((lane & 16) ? 8 : 0);
                int nl = ncol + t * 16 + (lane & 15);
                T[nl * TT_STRIDE + ml] = (__bf16)(acc[t][r] + bias[m0 + ml]);
            }
        }
        __syncthreads();
        __bf16* G = (part == 0) ? Qp : Kp;
        int l   = tid >> 1;               // 0..127
        int mmb = (tid & 1) * 32;
        size_t base = ((size_t)(b * NH_ + hh) * L_ + n0 + l) * HD_ + d0;
#pragma unroll
        for (int i = 0; i < 4; ++i)
            *(uint4*)&G[base + mmb + i * 8] = *(uint4*)&T[l * TT_STRIDE + mmb + i * 8];
    }
}

// ---------------------------------------------------------------------------
// Kernel 4: fused attention, one workgroup per (b, h, 64-query block).
// Dynamic LDS (~184KB of 320KB WGP LDS):
//   Sb[64][1032] bf16 scores/probs, Qt[64][136], double-buffered K/V tiles.
// K tiles double-buffered in pass 1; V tiles double-buffered in pass 3 with
// the first V copy issued BEFORE softmax so it overlaps the exp/normalize.
// ---------------------------------------------------------------------------
#define SB_STRIDE 1032
#define QT_STRIDE 136
#define VT_STRIDE 72
#define KB_ELEMS (64 * QT_STRIDE)     // 8704
#define VB_ELEMS (128 * VT_STRIDE)    // 9216
#define SMEM_SB  (64 * SB_STRIDE)
#define SMEM_QT  (64 * QT_STRIDE)
#define SMEM_KV  (2 * VB_ELEMS)       // 18432 >= 2*KB_ELEMS
#define ATTN_SMEM_BYTES ((SMEM_SB + SMEM_QT + SMEM_KV) * 2 + 512 * 4)

__global__ void attn_kernel(const __bf16* __restrict__ Qg,  // [b][h][L][D]
                            const __bf16* __restrict__ Kg,  // [b][h][L][D]
                            const __bf16* __restrict__ Vg,  // [b][h][D][L]
                            __bf16* __restrict__ AO) {      // [b][L][C]
    extern __shared__ char smem_raw[];
    __bf16* Sb  = (__bf16*)smem_raw;
    __bf16* Qt  = Sb + SMEM_SB;
    __bf16* KVt = Qt + SMEM_QT;
    float*  red = (float*)(KVt + SMEM_KV);

    const int qb = blockIdx.x, h = blockIdx.y, b = blockIdx.z;
    const int tid = threadIdx.x, lane = tid & 31, wv = tid >> 5;
    const size_t headBase = (size_t)(b * NH_ + h) * HD_ * L_;
    const float scale = 0.088388347648318440550f;           // 1/sqrt(128)

    auto stageK = [&](int buf, int n0) {
        __bf16* Kb = KVt + buf * KB_ELEMS;
#pragma unroll
        for (int i = 0; i < 4; ++i) {
            int idx = tid + i * 256;
            int m = idx >> 4, d8 = (idx & 15) * 8;
            cp_b128(&Kb[m * QT_STRIDE + d8],
                    &Kg[headBase + (size_t)(n0 + m) * HD_ + d8]);
        }
    };
    auto stageV = [&](int buf, int m0) {
        __bf16* Vb = KVt + buf * VB_ELEMS;
#pragma unroll
        for (int i = 0; i < 4; ++i) {
            int idx = tid + i * 256;
            int d = idx >> 3, m8 = (idx & 7) * 8;
            cp_b128(&Vb[d * VT_STRIDE + m8],
                    &Vg[headBase + (size_t)d * L_ + m0 + m8]);
        }
    };

    // stage Q tile (read every pass-1 iteration) + first K tile
#pragma unroll
    for (int i = 0; i < 4; ++i) {
        int idx = tid + i * 256;
        int l = idx >> 4, d8 = (idx & 15) * 8;
        cp_b128(&Qt[l * QT_STRIDE + d8],
                &Qg[headBase + (size_t)(qb * 64 + l) * HD_ + d8]);
    }
    stageK(0, 0);

    // ---- pass 1: S = scale * Q K^T, 64 x 1024, double-buffered K ----------
    {
        const int mrow = (wv & 3) * 16;
        const int ncol = (wv >> 2) * 32;
        for (int blk = 0; blk < 16; ++blk) {
            const int n0 = blk * 64, cur = blk & 1;
            if (blk + 1 < 16) { stageK(1 - cur, n0 + 64); CP_WAIT(0x4); }
            else              { CP_WAIT(0x0); }
            __syncthreads();
            const __bf16* Kb = KVt + cur * KB_ELEMS;
            v8f acc0 = {0.f,0.f,0.f,0.f,0.f,0.f,0.f,0.f};
            v8f acc1 = {0.f,0.f,0.f,0.f,0.f,0.f,0.f,0.f};
#pragma unroll
            for (int d0 = 0; d0 < HD_; d0 += 32) {
                v16bf a  = ld_frag(Qt, mrow + (lane & 15), QT_STRIDE, d0, lane);
                v16bf b0 = ld_frag(Kb, ncol + (lane & 15), QT_STRIDE, d0, lane);
                v16bf b1 = ld_frag(Kb, ncol + 16 + (lane & 15), QT_STRIDE, d0, lane);
                acc0 = wmma_bf16(a, b0, acc0);
                acc1 = wmma_bf16(a, b1, acc1);
            }
#pragma unroll
            for (int r = 0; r < 8; ++r) {
                int row = mrow + r + ((lane & 16) ? 8 : 0);
                int nn  = ncol + (lane & 15);
                Sb[row * SB_STRIDE + n0 + nn]      = (__bf16)(acc0[r] * scale);
                Sb[row * SB_STRIDE + n0 + nn + 16] = (__bf16)(acc1[r] * scale);
            }
            __syncthreads();
        }
    }

    // first V tile copy overlaps the softmax pass
    stageV(0, 0);

    // ---- pass 2: row softmax, 4 threads/row, uint4 (8x bf16) LDS ops ------
    {
        const int row = tid & 63, qtr = tid >> 6;
        __bf16* rp = Sb + row * SB_STRIDE + qtr * 256;
        float mx = -3.0e38f;
        for (int j = 0; j < 32; ++j) {
            Pack8 t; t.u = *(uint4*)&rp[j * 8];
#pragma unroll
            for (int k = 0; k < 8; ++k) mx = fmaxf(mx, (float)t.h[k]);
        }
        red[row * 4 + qtr] = mx;
        __syncthreads();
        mx = fmaxf(fmaxf(red[row * 4 + 0], red[row * 4 + 1]),
                   fmaxf(red[row * 4 + 2], red[row * 4 + 3]));
        float sum = 0.f;
        for (int j = 0; j < 32; ++j) {
            Pack8 t; t.u = *(uint4*)&rp[j * 8];
#pragma unroll
            for (int k = 0; k < 8; ++k) {
                float e = __expf((float)t.h[k] - mx);
                t.h[k] = (__bf16)e;
                sum += e;
            }
            *(uint4*)&rp[j * 8] = t.u;
        }
        red[256 + row * 4 + qtr] = sum;
        __syncthreads();
        float tot = red[256 + row * 4 + 0] + red[256 + row * 4 + 1] +
                    red[256 + row * 4 + 2] + red[256 + row * 4 + 3];
        float invs = 1.0f / tot;
        for (int j = 0; j < 32; ++j) {
            Pack8 t; t.u = *(uint4*)&rp[j * 8];
#pragma unroll
            for (int k = 0; k < 8; ++k) t.h[k] = (__bf16)((float)t.h[k] * invs);
            *(uint4*)&rp[j * 8] = t.u;
        }
        __syncthreads();
    }

    // ---- pass 3: O = P @ V^T (64 x 128), double-buffered V ----------------
    {
        const int mrow = (wv & 3) * 16;
        const int dcol = (wv >> 2) * 64;
        v8f acc[4];
#pragma unroll
        for (int t = 0; t < 4; ++t) acc[t] = (v8f){0.f,0.f,0.f,0.f,0.f,0.f,0.f,0.f};

        for (int blk = 0; blk < 16; ++blk) {
            const int m0 = blk * 64, cur = blk & 1;
            if (blk + 1 < 16) { stageV(1 - cur, m0 + 64); CP_WAIT(0x4); }
            else              { CP_WAIT(0x0); }
            __syncthreads();
            const __bf16* Vb = KVt + cur * VB_ELEMS;
#pragma unroll
            for (int k0 = 0; k0 < 64; k0 += 32) {
                v16bf a = ld_frag(Sb, mrow + (lane & 15), SB_STRIDE, m0 + k0, lane);
#pragma unroll
                for (int t = 0; t < 4; ++t) {
                    v16bf bf = ld_frag(Vb, dcol + t * 16 + (lane & 15),
                                       VT_STRIDE, k0, lane);
                    acc[t] = wmma_bf16(a, bf, acc[t]);
                }
            }
            __syncthreads();
        }
        // AO [b][L][C]: lanes contiguous along c -> coalesced
#pragma unroll
        for (int t = 0; t < 4; ++t) {
#pragma unroll
            for (int r = 0; r < 8; ++r) {
                int d = dcol + t * 16 + (lane & 15);
                int l = qb * 64 + mrow + r + ((lane & 16) ? 8 : 0);
                AO[((size_t)b * L_ + l) * C_ + h * HD_ + d] = (__bf16)acc[t][r];
            }
        }
    }
}

// ---------------------------------------------------------------------------
// Host launcher
// ---------------------------------------------------------------------------
extern "C" void kernel_launch(void* const* d_in, const int* in_sizes, int n_in,
                              void* d_out, int out_size, void* d_ws, size_t ws_size,
                              hipStream_t stream) {
    const float* x      = (const float*)d_in[0];
    const float* gamma  = (const float*)d_in[1];
    const float* beta   = (const float*)d_in[2];
    const float* w_qkv  = (const float*)d_in[3];
    const float* b_qkv  = (const float*)d_in[4];
    const float* w_proj = (const float*)d_in[5];
    const float* b_proj = (const float*)d_in[6];
    float* out = (float*)d_out;

    char* ws = (char*)d_ws;
    const size_t bclL = (size_t)B_ * C_ * L_ * sizeof(__bf16);   // 16.78 MB
    __bf16* XN = (__bf16*)ws;                       ws += bclL;  // [b][L][C]
    __bf16* WQ = (__bf16*)ws;                       ws += (size_t)3 * C_ * C_ * 2;
    __bf16* WP = (__bf16*)ws;                       ws += (size_t)C_ * C_ * 2;
    __bf16* Q  = (__bf16*)ws;                       ws += bclL;  // [b][h][L][D]
    __bf16* K  = (__bf16*)ws;                       ws += bclL;  // [b][h][L][D]
    __bf16* V  = (__bf16*)ws;                       ws += bclL;  // [b][h][D][L]
    __bf16* AO = (__bf16*)ws;                       ws += bclL;  // [b][L][C]

    f32_to_bf16_kernel<<<(3 * C_ * C_ + 255) / 256, 256, 0, stream>>>(w_qkv, WQ, 3 * C_ * C_);
    f32_to_bf16_kernel<<<(C_ * C_ + 255) / 256, 256, 0, stream>>>(w_proj, WP, C_ * C_);

    groupnorm_kernel<<<dim3(NG_, B_), 256, 0, stream>>>(x, gamma, beta, XN);

    gemm_kernel<<<dim3(L_ / 128, (3 * C_) / 64, B_), 256, 0, stream>>>(
        WQ, XN, b_qkv, Q, K, V, nullptr, nullptr, 0);

    attn_kernel<<<dim3(L_ / 64, NH_, B_), 256, ATTN_SMEM_BYTES, stream>>>(Q, K, V, AO);

    gemm_kernel<<<dim3(L_ / 128, C_ / 64, B_), 256, 0, stream>>>(
        WP, AO, b_proj, nullptr, nullptr, nullptr, out, x, 1);
}
